// GN_NN_32873679684148
// MI455X (gfx1250) — compile-verified, matrix operand order checked by reading, not run on hardware
//
#include <hip/hip_runtime.h>
#include <math.h>

// ---------------------------------------------------------------------------
// GNN (encoder MLP+LN, 4x GATv2, decoder MLP) for MI455X / gfx1250.
// - All matmuls on v_wmma_f32_16x16x32_bf16 (bf16 in, f32 accum).
// - Weight panels DMA'd into LDS with the Tensor Data Mover
//   (tensor_load_to_lds + s_wait_tensorcnt).
// - global_prefetch for the irregular xl[src]/xr[dst] gathers.
// Workspace requirement: ~350 MB (assumed available in d_ws).
// ---------------------------------------------------------------------------

typedef __attribute__((ext_vector_type(16))) __bf16 v16bf;
typedef __attribute__((ext_vector_type(8)))  __bf16 v8bf;
typedef __attribute__((ext_vector_type(8)))  float  v8f;
typedef unsigned int u32x4 __attribute__((ext_vector_type(4)));
typedef int          i32x4 __attribute__((ext_vector_type(4)));
typedef int          i32x8 __attribute__((ext_vector_type(8)));

#define N_NODES 50000
#define N_EDGES 800000
#define HID     128
#define NLAYERS 4
#define HH      (HID * HID)

// ---------------- helpers ----------------

__device__ __forceinline__ __bf16 f2bf(float x) {
  unsigned u = __float_as_uint(x);
  unsigned r = (u + 0x7FFFu + ((u >> 16) & 1u)) >> 16;   // RNE
  unsigned short h = (unsigned short)r;
  __bf16 b;
  __builtin_memcpy(&b, &h, 2);
  return b;
}

__device__ __forceinline__ float lrelu(float x, float a) {
  return x > 0.f ? x : a * x;
}

// Load a 16x32 bf16 A/B fragment (row-major source, `stride` in elements).
// ISA 7.12.2 layout: lane<16 -> K {0..7,16..23}; lane>=16 -> K {8..15,24..31}.
__device__ __forceinline__ v16bf load_frag(const __bf16* base, int stride,
                                           int lane, int k0) {
  const __bf16* p = base + (size_t)(lane & 15) * stride + k0 + ((lane >> 4) << 3);
  v8bf lo = *(const v8bf*)(p);
  v8bf hi = *(const v8bf*)(p + 16);
  return __builtin_shufflevector(lo, hi, 0, 1, 2, 3, 4, 5, 6, 7,
                                 8, 9, 10, 11, 12, 13, 14, 15);
}

__device__ __forceinline__ v8f wmma_bf16(v16bf a, v16bf b, v8f c) {
  return __builtin_amdgcn_wmma_f32_16x16x32_bf16(false, a, false, b,
                                                 (short)0, c, false, false);
}

__device__ __forceinline__ void atomicMaxFloat(float* addr, float val) {
  unsigned* ua = (unsigned*)addr;
  unsigned old = __float_as_uint(*addr);
  while (__uint_as_float(old) < val) {
    unsigned assumed = old;
    old = atomicCAS(ua, assumed, __float_as_uint(val));
    if (old == assumed) break;
  }
}

// Low 32 bits of a flat shared-memory address == LDS byte offset (aperture
// layout, ISA 10.2).
__device__ __forceinline__ unsigned lds_off(const void* p) {
  return (unsigned)(unsigned long long)(uintptr_t)p;
}

// TDM: contiguous 1-D copy global -> LDS (nbytes multiple of 8, <= 512 KB).
// D# group0: count=1, lds_addr, 57b global addr, type=2 ("image").
// D# group1: data_size=8B, tensor_dim0=tile_dim0=nbytes/8, tensor_dim1=1.
__device__ __forceinline__ void tdm_load_1d(unsigned lds_addr, const void* gptr,
                                            unsigned nbytes) {
  unsigned long long ga = (unsigned long long)(uintptr_t)gptr;
  unsigned n8 = nbytes >> 3;
  u32x4 g0;
  g0[0] = 1u;                                             // count=1 (valid)
  g0[1] = lds_addr;                                       // lds_addr[31:0]
  g0[2] = (unsigned)(ga & 0xffffffffu);                   // global_addr lo
  g0[3] = (unsigned)((ga >> 32) & 0x01ffffffu) | (2u << 30);  // addr hi | type=2
  i32x8 g1;
  g1[0] = (int)(3u << 16);                    // data_size=3 (8B), mask=0
  g1[1] = (int)((n8 & 0xffffu) << 16);        // tensor_dim0[15:0]
  g1[2] = (int)(((n8 >> 16) & 0xffffu) | (1u << 16));  // td0 hi | tensor_dim1=1
  g1[3] = (int)((n8 & 0xffffu) << 16);        // td1 hi(0) | tile_dim0
  g1[4] = 1;                                  // tile_dim1=1, tile_dim2=0
  g1[5] = (int)n8;                            // tensor_dim0_stride lo
  g1[6] = 0;                                  // stride hi | dim1_stride lo
  g1[7] = 0;
  i32x4 z4 = {0, 0, 0, 0};
#if __clang_major__ >= 23
  i32x8 z8 = {0, 0, 0, 0, 0, 0, 0, 0};
  __builtin_amdgcn_tensor_load_to_lds(g0, g1, z4, z4, z8, 0);
#else
  __builtin_amdgcn_tensor_load_to_lds(g0, g1, z4, z4, 0);
#endif
}

__device__ __forceinline__ void tdm_wait() {
  __builtin_amdgcn_s_wait_tensorcnt((short)0);
}

// ---------------- weight prep (fp32 -> bf16, with padding) -----------------

__global__ void convert3_kernel(const float* __restrict__ a,
                                const float* __restrict__ b,
                                const float* __restrict__ c,
                                __bf16* __restrict__ oa,
                                __bf16* __restrict__ ob,
                                __bf16* __restrict__ oc, int n) {
  int i = blockIdx.x * blockDim.x + threadIdx.x;
  if (i < n) {
    oa[i] = f2bf(a[i]);
    ob[i] = f2bf(b[i]);
    oc[i] = f2bf(c[i]);
  }
}

__global__ void prep_node_enc_w_kernel(const float* __restrict__ w1,
                                       const float* __restrict__ w2,
                                       const float* __restrict__ w3,
                                       __bf16* __restrict__ o1,
                                       __bf16* __restrict__ o2,
                                       __bf16* __restrict__ o3) {
  int i = blockIdx.x * blockDim.x + threadIdx.x;
  if (i < 64 * 32) o1[i] = f2bf(w1[i]);
  if (i < 96 * 64) o2[i] = f2bf(w2[i]);
  if (i < 128 * 96) o3[i] = f2bf(w3[i]);
}

__global__ void prep_edge_enc_w_kernel(const float* __restrict__ w1,
                                       const float* __restrict__ w2,
                                       const float* __restrict__ w3,
                                       __bf16* __restrict__ o1,
                                       __bf16* __restrict__ o2,
                                       __bf16* __restrict__ o3) {
  int i = blockIdx.x * blockDim.x + threadIdx.x;
  if (i < 64 * 32) {   // [48][8] -> [64][32]
    int j = i >> 5, k = i & 31;
    o1[i] = f2bf((j < 48 && k < 8) ? w1[j * 8 + k] : 0.f);
  }
  if (i < 96 * 64) {   // [88][48] -> [96][64]
    int j = i >> 6, k = i & 63;
    o2[i] = f2bf((j < 88 && k < 48) ? w2[j * 48 + k] : 0.f);
  }
  if (i < 128 * 96) {  // [128][88] -> [128][96]
    int j = i / 96, k = i % 96;
    o3[i] = f2bf((k < 88) ? w3[j * 88 + k] : 0.f);
  }
}

__global__ void prep_dec_w1_kernel(const float* __restrict__ w1,
                                   __bf16* __restrict__ o) {
  int i = blockIdx.x * blockDim.x + threadIdx.x;
  if (i < 96 * 128) {  // [86][128] -> [96][128]
    int j = i >> 7, k = i & 127;
    o[i] = f2bf(j < 86 ? w1[j * 128 + k] : 0.f);
  }
}

__global__ void init_buffers_kernel(float* __restrict__ outacc,
                                    float* __restrict__ mbuf,
                                    float* __restrict__ dbuf,
                                    int total, int nnodes) {
  int i = blockIdx.x * blockDim.x + threadIdx.x;
  if (i < total) outacc[i] = 0.f;
  if (i < nnodes) { mbuf[i] = -3.0e38f; dbuf[i] = 0.f; }
}

// ---------------- node encoder: 32 -> 64 -> 96 -> 128 + LN ----------------
// 32 rows per block, 8 waves; weights DMA'd per stage via TDM.

__global__ void node_encoder_kernel(
    const float* __restrict__ x,
    const __bf16* __restrict__ w1bf, const float* __restrict__ b1,
    const __bf16* __restrict__ w2bf, const float* __restrict__ b2,
    const __bf16* __restrict__ w3bf, const float* __restrict__ b3,
    const float* __restrict__ lng, const float* __restrict__ lnb,
    float* __restrict__ y, __bf16* __restrict__ ybf, int M) {
  extern __shared__ char smem[];
  __bf16* sx   = (__bf16*)smem;            // [32][32]
  __bf16* sh1  = sx + 32 * 32;             // [32][64]
  __bf16* sh2  = sh1 + 32 * 64;            // [32][96]
  __bf16* sw   = sh2 + 32 * 96;            // up to [128][96]
  float*  sb   = (float*)(sw + 128 * 96);  // [128]
  float*  sout = sb + 128;                 // [32][128]
  float*  smu  = sout + 32 * 128;          // [32]
  float*  srs  = smu + 32;                 // [32]

  const int tid = threadIdx.x;
  const int wave = tid >> 5, lane = tid & 31;
  const int r0 = blockIdx.x * 32;

  for (int i = tid; i < 32 * 32; i += 256) {
    int r = i >> 5, c = i & 31, g = r0 + r;
    sx[i] = f2bf(g < M ? x[(size_t)g * 32 + c] : 0.f);
  }
  if (tid < 32) { tdm_load_1d(lds_off(sw), w1bf, 64 * 32 * 2); tdm_wait(); }
  for (int i = tid; i < 64; i += 256) sb[i] = b1[i];
  __syncthreads();

  {  // layer1: [32][64], K=32, 8 tiles -> 1/wave
    int mt = wave >> 2, nt = wave & 3;
    v8f acc = {};
    v16bf a = load_frag(sx + mt * 16 * 32, 32, lane, 0);
    v16bf b = load_frag(sw + nt * 16 * 32, 32, lane, 0);
    acc = wmma_bf16(a, b, acc);
    int col = nt * 16 + (lane & 15);
    int rbase = mt * 16 + ((lane >> 4) << 3);
    for (int v = 0; v < 8; v++)
      sh1[(size_t)(rbase + v) * 64 + col] = f2bf(lrelu(acc[v] + sb[col], 0.01f));
  }
  __syncthreads();
  if (tid < 32) { tdm_load_1d(lds_off(sw), w2bf, 96 * 64 * 2); tdm_wait(); }
  for (int i = tid; i < 96; i += 256) sb[i] = b2[i];
  __syncthreads();

  for (int t = wave; t < 12; t += 8) {  // layer2: [32][96], K=64
    int mt = t / 6, nt = t % 6;
    v8f acc = {};
    for (int k0 = 0; k0 < 64; k0 += 32) {
      v16bf a = load_frag(sh1 + mt * 16 * 64, 64, lane, k0);
      v16bf b = load_frag(sw + nt * 16 * 64, 64, lane, k0);
      acc = wmma_bf16(a, b, acc);
    }
    int col = nt * 16 + (lane & 15);
    int rbase = mt * 16 + ((lane >> 4) << 3);
    for (int v = 0; v < 8; v++)
      sh2[(size_t)(rbase + v) * 96 + col] = f2bf(lrelu(acc[v] + sb[col], 0.01f));
  }
  __syncthreads();
  if (tid < 32) { tdm_load_1d(lds_off(sw), w3bf, 128 * 96 * 2); tdm_wait(); }
  for (int i = tid; i < 128; i += 256) sb[i] = b3[i];
  __syncthreads();

  for (int t = wave; t < 16; t += 8) {  // layer3: [32][128], K=96 (linear)
    int mt = t >> 3, nt = t & 7;
    v8f acc = {};
    for (int k0 = 0; k0 < 96; k0 += 32) {
      v16bf a = load_frag(sh2 + mt * 16 * 96, 96, lane, k0);
      v16bf b = load_frag(sw + nt * 16 * 96, 96, lane, k0);
      acc = wmma_bf16(a, b, acc);
    }
    int col = nt * 16 + (lane & 15);
    int rbase = mt * 16 + ((lane >> 4) << 3);
    for (int v = 0; v < 8; v++)
      sout[(size_t)(rbase + v) * 128 + col] = acc[v] + sb[col];
  }
  __syncthreads();

  if (tid < 32) {
    float s = 0.f, ss = 0.f;
    for (int c = 0; c < 128; c++) {
      float v = sout[tid * 128 + c];
      s += v; ss += v * v;
    }
    float mu = s * (1.f / 128.f);
    smu[tid] = mu;
    srs[tid] = rsqrtf(ss * (1.f / 128.f) - mu * mu + 1e-5f);
  }
  __syncthreads();
  for (int i = tid; i < 32 * 128; i += 256) {
    int r = i >> 7, c = i & 127, g = r0 + r;
    if (g < M) {
      float val = (sout[i] - smu[r]) * srs[r] * lng[c] + lnb[c];
      y[(size_t)g * 128 + c] = val;
      ybf[(size_t)g * 128 + c] = f2bf(val);
    }
  }
}

// ---------------- edge encoder: 8 -> 48 -> 88 -> 128 + LN ------------------
// Same structure; weights are pre-padded bf16 panels, DMA'd via TDM.

__global__ void edge_encoder_kernel(
    const float* __restrict__ ea,
    const __bf16* __restrict__ w1bf, const float* __restrict__ b1,
    const __bf16* __restrict__ w2bf, const float* __restrict__ b2,
    const __bf16* __restrict__ w3bf, const float* __restrict__ b3,
    const float* __restrict__ lng, const float* __restrict__ lnb,
    __bf16* __restrict__ eabf, int M) {
  extern __shared__ char smem[];
  __bf16* sx   = (__bf16*)smem;            // [32][32] (cols>=8 zero)
  __bf16* sh1  = sx + 32 * 32;             // [32][64]
  __bf16* sh2  = sh1 + 32 * 64;            // [32][96]
  __bf16* sw   = sh2 + 32 * 96;            // up to [128][96]
  float*  sb   = (float*)(sw + 128 * 96);
  float*  sout = sb + 128;                 // [32][128]
  float*  smu  = sout + 32 * 128;
  float*  srs  = smu + 32;

  const int tid = threadIdx.x;
  const int wave = tid >> 5, lane = tid & 31;
  const int r0 = blockIdx.x * 32;

  for (int i = tid; i < 32 * 32; i += 256) {
    int r = i >> 5, c = i & 31, g = r0 + r;
    sx[i] = f2bf((g < M && c < 8) ? ea[(size_t)g * 8 + c] : 0.f);
  }
  if (tid < 32) { tdm_load_1d(lds_off(sw), w1bf, 64 * 32 * 2); tdm_wait(); }
  for (int i = tid; i < 64; i += 256) sb[i] = (i < 48) ? b1[i] : 0.f;
  __syncthreads();

  {  // layer1: [32][64], K=32
    int mt = wave >> 2, nt = wave & 3;
    v8f acc = {};
    v16bf a = load_frag(sx + mt * 16 * 32, 32, lane, 0);
    v16bf b = load_frag(sw + nt * 16 * 32, 32, lane, 0);
    acc = wmma_bf16(a, b, acc);
    int col = nt * 16 + (lane & 15);
    int rbase = mt * 16 + ((lane >> 4) << 3);
    for (int v = 0; v < 8; v++)
      sh1[(size_t)(rbase + v) * 64 + col] = f2bf(lrelu(acc[v] + sb[col], 0.01f));
  }
  __syncthreads();
  if (tid < 32) { tdm_load_1d(lds_off(sw), w2bf, 96 * 64 * 2); tdm_wait(); }
  for (int i = tid; i < 96; i += 256) sb[i] = (i < 88) ? b2[i] : 0.f;
  __syncthreads();

  for (int t = wave; t < 12; t += 8) {  // layer2: [32][96], K=64
    int mt = t / 6, nt = t % 6;
    v8f acc = {};
    for (int k0 = 0; k0 < 64; k0 += 32) {
      v16bf a = load_frag(sh1 + mt * 16 * 64, 64, lane, k0);
      v16bf b = load_frag(sw + nt * 16 * 64, 64, lane, k0);
      acc = wmma_bf16(a, b, acc);
    }
    int col = nt * 16 + (lane & 15);
    int rbase = mt * 16 + ((lane >> 4) << 3);
    for (int v = 0; v < 8; v++)
      sh2[(size_t)(rbase + v) * 96 + col] = f2bf(lrelu(acc[v] + sb[col], 0.01f));
  }
  __syncthreads();
  if (tid < 32) { tdm_load_1d(lds_off(sw), w3bf, 128 * 96 * 2); tdm_wait(); }
  for (int i = tid; i < 128; i += 256) sb[i] = b3[i];
  __syncthreads();

  for (int t = wave; t < 16; t += 8) {  // layer3: [32][128], K=96
    int mt = t >> 3, nt = t & 7;
    v8f acc = {};
    for (int k0 = 0; k0 < 96; k0 += 32) {
      v16bf a = load_frag(sh2 + mt * 16 * 96, 96, lane, k0);
      v16bf b = load_frag(sw + nt * 16 * 96, 96, lane, k0);
      acc = wmma_bf16(a, b, acc);
    }
    int col = nt * 16 + (lane & 15);
    int rbase = mt * 16 + ((lane >> 4) << 3);
    for (int v = 0; v < 8; v++)
      sout[(size_t)(rbase + v) * 128 + col] = acc[v] + sb[col];
  }
  __syncthreads();

  if (tid < 32) {
    float s = 0.f, ss = 0.f;
    for (int c = 0; c < 128; c++) {
      float v = sout[tid * 128 + c];
      s += v; ss += v * v;
    }
    float mu = s * (1.f / 128.f);
    smu[tid] = mu;
    srs[tid] = rsqrtf(ss * (1.f / 128.f) - mu * mu + 1e-5f);
  }
  __syncthreads();
  for (int i = tid; i < 32 * 128; i += 256) {
    int r = i >> 7, c = i & 127, g = r0 + r;
    if (g < M)
      eabf[(size_t)g * 128 + c] =
          f2bf((sout[i] - smu[r]) * srs[r] * lng[c] + lnb[c]);
  }
}

// ---------------- 128x128 GEMM: C = A(bf16) * W^T + bias -------------------
// 80 rows per block (50000 = 625*80), B panel (32 KB) TDM'd into LDS once.

__global__ void gemm128_kernel(const __bf16* __restrict__ A,
                               const __bf16* __restrict__ B,
                               const float* __restrict__ bias,
                               float* __restrict__ C) {
  __shared__ __align__(16) __bf16 sB[128 * 128];
  const int tid = threadIdx.x;
  const int wave = tid >> 5, lane = tid & 31;
  const int rowb = blockIdx.x * 80;
  if (tid < 32) { tdm_load_1d(lds_off(sB), B, 128 * 128 * 2); tdm_wait(); }
  __syncthreads();
  for (int t = wave; t < 40; t += 8) {  // 5 mtiles x 8 ntiles
    const int row0 = rowb + (t >> 3) * 16;
    const int n0 = (t & 7) * 16;
    v8f acc = {};
    for (int k0 = 0; k0 < 128; k0 += 32) {
      v16bf a = load_frag(A + (size_t)row0 * 128, 128, lane, k0);
      v16bf b = load_frag(sB + n0 * 128, 128, lane, k0);
      acc = wmma_bf16(a, b, acc);
    }
    const int col = n0 + (lane & 15);
    const int rbase = row0 + ((lane >> 4) << 3);
    const float bi = bias[col];
    for (int v = 0; v < 8; v++)
      C[(size_t)(rbase + v) * 128 + col] = acc[v] + bi;
  }
}

// ---------------- fused edge-logit pass ------------------------------------
// 64 edges / block: eW = ea @ we^T (WMMA, weights TDM'd to LDS),
// e = eW + xl[src] + xr[dst] (rows prefetched), s = leaky(e,0.2).att,
// atomic segment-max into mbuf[dst].

__global__ void edge_logits_kernel(const __bf16* __restrict__ eabf,
                                   const __bf16* __restrict__ weB,
                                   const float* __restrict__ att,
                                   const float* __restrict__ xl,
                                   const float* __restrict__ xr,
                                   const int* __restrict__ src,
                                   const int* __restrict__ dst,
                                   float* __restrict__ sout,
                                   float* __restrict__ mbuf) {
  __shared__ __align__(16) __bf16 sW[128 * 128];
  __shared__ float s_sum[64];
  const int tid = threadIdx.x;
  const int wave = tid >> 5, lane = tid & 31;
  const int e0 = blockIdx.x * 64;

  if (tid < 64) {                     // prefetch gather rows (512 B each)
    const float* p = xl + (size_t)src[e0 + tid] * 128;
    __builtin_prefetch(p, 0, 1);
    __builtin_prefetch(p + 64, 0, 1);
    s_sum[tid] = 0.f;
  } else if (tid < 128) {
    const float* p = xr + (size_t)dst[e0 + tid - 64] * 128;
    __builtin_prefetch(p, 0, 1);
    __builtin_prefetch(p + 64, 0, 1);
  }
  if (tid < 32) { tdm_load_1d(lds_off(sW), weB, 128 * 128 * 2); tdm_wait(); }
  __syncthreads();

  for (int t = wave; t < 32; t += 8) {  // 4 edge-tiles x 8 col-tiles
    const int eb = e0 + (t >> 3) * 16;
    const int n0 = (t & 7) * 16;
    v8f acc = {};
    for (int k0 = 0; k0 < 128; k0 += 32) {
      v16bf a = load_frag(eabf + (size_t)eb * 128, 128, lane, k0);
      v16bf b = load_frag(sW + n0 * 128, 128, lane, k0);
      acc = wmma_bf16(a, b, acc);
    }
    const int col = n0 + (lane & 15);
    const int half8 = (lane >> 4) << 3;
    const float attc = att[col];
    for (int v = 0; v < 8; v++) {
      int m = v + half8;
      int e = eb + m;
      int sv = src[e], dv = dst[e];
      float val = acc[v] + xl[(size_t)sv * 128 + col] + xr[(size_t)dv * 128 + col];
      float partial = lrelu(val, 0.2f) * attc;
      partial += __shfl_xor(partial, 1, 16);
      partial += __shfl_xor(partial, 2, 16);
      partial += __shfl_xor(partial, 4, 16);
      partial += __shfl_xor(partial, 8, 16);
      if ((lane & 15) == 0) atomicAdd(&s_sum[(eb - e0) + m], partial);
    }
  }
  __syncthreads();
  if (tid < 64) {
    int e = e0 + tid;
    float s = s_sum[tid];
    sout[e] = s;
    atomicMaxFloat(&mbuf[dst[e]], s);
  }
}

// ---------------- softmax normalization over edges -------------------------

__global__ void edge_softmax_kernel(const float* __restrict__ s,
                                    const int* __restrict__ dst,
                                    const float* __restrict__ mbuf,
                                    float* __restrict__ dbuf,
                                    float* __restrict__ z, int E) {
  int e = blockIdx.x * blockDim.x + threadIdx.x;
  if (e >= E) return;
  int d = dst[e];
  float zv = __expf(s[e] - mbuf[d]);
  z[e] = zv;
  atomicAdd(&dbuf[d], zv);
}

__global__ void edge_scatter_kernel(const float* __restrict__ z,
                                    const float* __restrict__ dbuf,
                                    const float* __restrict__ xl,
                                    const int* __restrict__ src,
                                    const int* __restrict__ dst,
                                    float* __restrict__ outacc, int total) {
  int idx = blockIdx.x * blockDim.x + threadIdx.x;
  if (idx >= total) return;
  int e = idx >> 7, c = idx & 127;
  int d = dst[e];
  float alpha = z[e] / fmaxf(dbuf[d], 1e-16f);
  atomicAdd(&outacc[(size_t)d * 128 + c],
            alpha * xl[(size_t)src[e] * 128 + c]);
}

// ---------------- per-node: y += LN(out + bias); reset accumulators --------

__global__ void node_update_kernel(float* __restrict__ y,
                                   __bf16* __restrict__ ybf,
                                   float* __restrict__ outacc,
                                   const float* __restrict__ bias,
                                   const float* __restrict__ lng,
                                   const float* __restrict__ lnb,
                                   float* __restrict__ mbuf,
                                   float* __restrict__ dbuf, int M) {
  int n = blockIdx.x * blockDim.x + threadIdx.x;
  if (n >= M) return;
  float* orow = outacc + (size_t)n * 128;
  float s = 0.f, ss = 0.f;
  for (int c = 0; c < 128; c++) {
    float g = orow[c] + bias[c];
    s += g; ss += g * g;
  }
  float mu = s * (1.f / 128.f);
  float rs = rsqrtf(ss * (1.f / 128.f) - mu * mu + 1e-5f);
  for (int c = 0; c < 128; c++) {
    float g = orow[c] + bias[c];
    float v = (g - mu) * rs * lng[c] + lnb[c];
    float ny = y[(size_t)n * 128 + c] + v;
    y[(size_t)n * 128 + c] = ny;
    ybf[(size_t)n * 128 + c] = f2bf(ny);
    orow[c] = 0.f;  // ready for next layer
  }
  mbuf[n] = -3.0e38f;
  dbuf[n] = 0.f;
}

// ---------------- decoder layer1 (128 -> 86, padded to 96), WMMA -----------
// 80 rows per block; pre-padded bf16 weight panel (24 KB) TDM'd into LDS.

__global__ void dec1_kernel(const __bf16* __restrict__ ybf,
                            const __bf16* __restrict__ w1bf,
                            const float* __restrict__ b1,
                            float* __restrict__ h1) {
  __shared__ __align__(16) __bf16 sW[96 * 128];
  __shared__ float sb[96];
  const int tid = threadIdx.x;
  const int wave = tid >> 5, lane = tid & 31;
  if (tid < 32) { tdm_load_1d(lds_off(sW), w1bf, 96 * 128 * 2); tdm_wait(); }
  for (int i = tid; i < 96; i += 256) sb[i] = (i < 86) ? b1[i] : 0.f;
  __syncthreads();

  const int rowb = blockIdx.x * 80;
  for (int t = wave; t < 30; t += 8) {  // 5 mtiles x 6 ntiles
    const int row0 = rowb + (t / 6) * 16;
    const int n0 = (t % 6) * 16;
    v8f acc = {};
    for (int k0 = 0; k0 < 128; k0 += 32) {
      v16bf a = load_frag(ybf + (size_t)row0 * 128, 128, lane, k0);
      v16bf b = load_frag(sW + n0 * 128, 128, lane, k0);
      acc = wmma_bf16(a, b, acc);
    }
    const int col = n0 + (lane & 15);
    const int rbase = row0 + ((lane >> 4) << 3);
    for (int v = 0; v < 8; v++)
      h1[(size_t)(rbase + v) * 96 + col] = lrelu(acc[v] + sb[col], 0.01f);
  }
}

// ---------------- decoder layers 2+3 (86 -> 44 -> 2), scalar ---------------

__global__ void dec23_kernel(const float* __restrict__ h1,
                             const float* __restrict__ w2,
                             const float* __restrict__ b2,
                             const float* __restrict__ w3,
                             const float* __restrict__ b3,
                             float* __restrict__ out, int M) {
  __shared__ float sw2[44 * 86];
  __shared__ float sb2[44];
  __shared__ float sw3[2 * 44];
  __shared__ float sb3[2];
  const int tid = threadIdx.x;
  for (int i = tid; i < 44 * 86; i += 256) sw2[i] = w2[i];
  for (int i = tid; i < 44; i += 256) sb2[i] = b2[i];
  for (int i = tid; i < 88; i += 256) sw3[i] = w3[i];
  for (int i = tid; i < 2; i += 256) sb3[i] = b3[i];
  __syncthreads();
  int n = blockIdx.x * 256 + tid;
  if (n >= M) return;
  float h2[44];
  for (int j = 0; j < 44; j++) h2[j] = sb2[j];
  const float* hr = h1 + (size_t)n * 96;
  for (int k = 0; k < 86; k++) {
    float v = hr[k];
    for (int j = 0; j < 44; j++) h2[j] += v * sw2[j * 86 + k];
  }
  for (int j = 0; j < 44; j++) h2[j] = lrelu(h2[j], 0.01f);
  for (int j = 0; j < 2; j++) {
    float o = sb3[j];
    for (int k = 0; k < 44; k++) o += h2[k] * sw3[j * 44 + k];
    out[(size_t)n * 2 + j] = o;
  }
}

// ---------------- host launcher --------------------------------------------

extern "C" void kernel_launch(void* const* d_in, const int* in_sizes, int n_in,
                              void* d_out, int out_size, void* d_ws,
                              size_t ws_size, hipStream_t stream) {
  (void)in_sizes; (void)n_in; (void)out_size; (void)ws_size;

  const float* x       = (const float*)d_in[0];
  const int*   ei      = (const int*)d_in[1];
  const float* eattr   = (const float*)d_in[2];
  const float* enc_w1  = (const float*)d_in[3];
  const float* enc_b1  = (const float*)d_in[4];
  const float* enc_w2  = (const float*)d_in[5];
  const float* enc_b2  = (const float*)d_in[6];
  const float* enc_w3  = (const float*)d_in[7];
  const float* enc_b3  = (const float*)d_in[8];
  const float* enc_lng = (const float*)d_in[9];
  const float* enc_lnb = (const float*)d_in[10];
  const float* ee_w1   = (const float*)d_in[11];
  const float* ee_b1   = (const float*)d_in[12];
  const float* ee_w2   = (const float*)d_in[13];
  const float* ee_b2   = (const float*)d_in[14];
  const float* ee_w3   = (const float*)d_in[15];
  const float* ee_b3   = (const float*)d_in[16];
  const float* ee_lng  = (const float*)d_in[17];
  const float* ee_lnb  = (const float*)d_in[18];
  const float* gat_wl  = (const float*)d_in[19];
  const float* gat_bl  = (const float*)d_in[20];
  const float* gat_wr  = (const float*)d_in[21];
  const float* gat_br  = (const float*)d_in[22];
  const float* gat_we  = (const float*)d_in[23];
  const float* gat_att = (const float*)d_in[24];
  const float* gat_bias= (const float*)d_in[25];
  const float* ln_g    = (const float*)d_in[26];
  const float* ln_b    = (const float*)d_in[27];
  const float* dec_w1  = (const float*)d_in[28];
  const float* dec_b1  = (const float*)d_in[29];
  const float* dec_w2  = (const float*)d_in[30];
  const float* dec_b2  = (const float*)d_in[31];
  const float* dec_w3  = (const float*)d_in[32];
  const float* dec_b3  = (const float*)d_in[33];

  const int* src = ei;
  const int* dst = ei + N_EDGES;

  // ---- workspace carving (≈350 MB total) ----
  char* p = (char*)d_ws;
  auto take = [&](size_t bytes) -> char* {
    char* r = p;
    p += (bytes + 255) & ~(size_t)255;
    return r;
  };
  float*  y      = (float*)take((size_t)N_NODES * HID * 4);
  __bf16* ybf    = (__bf16*)take((size_t)N_NODES * HID * 2);
  __bf16* eabf   = (__bf16*)take((size_t)N_EDGES * HID * 2);
  float*  xl     = (float*)take((size_t)N_NODES * HID * 4);
  float*  xr     = (float*)take((size_t)N_NODES * HID * 4);
  float*  outacc = (float*)take((size_t)N_NODES * HID * 4);
  float*  sbuf   = (float*)take((size_t)N_EDGES * 4);
  float*  zbuf   = (float*)take((size_t)N_EDGES * 4);
  float*  mbuf   = (float*)take((size_t)N_NODES * 4);
  float*  dbuf   = (float*)take((size_t)N_NODES * 4);
  __bf16* wlbf   = (__bf16*)take((size_t)NLAYERS * HH * 2);
  __bf16* wrbf   = (__bf16*)take((size_t)NLAYERS * HH * 2);
  __bf16* webf   = (__bf16*)take((size_t)NLAYERS * HH * 2);
  float*  h1dec  = (float*)take((size_t)N_NODES * 96 * 4);
  __bf16* ne_w1  = (__bf16*)take(64 * 32 * 2);
  __bf16* ne_w2  = (__bf16*)take(96 * 64 * 2);
  __bf16* ne_w3  = (__bf16*)take(128 * 96 * 2);
  __bf16* ee_w1p = (__bf16*)take(64 * 32 * 2);
  __bf16* ee_w2p = (__bf16*)take(96 * 64 * 2);
  __bf16* ee_w3p = (__bf16*)take(128 * 96 * 2);
  __bf16* dw1p   = (__bf16*)take(96 * 128 * 2);

  const size_t enc_smem =
      (32 * 32 + 32 * 64 + 32 * 96 + 128 * 96) * 2 +  // bf16 region
      (128 + 32 * 128 + 32 + 32) * 4;                 // f32 region  (~54 KB)

  // ---- prologue: weight prep ----
  convert3_kernel<<<(NLAYERS * HH + 255) / 256, 256, 0, stream>>>(
      gat_wl, gat_wr, gat_we, wlbf, wrbf, webf, NLAYERS * HH);
  prep_node_enc_w_kernel<<<(128 * 96 + 255) / 256, 256, 0, stream>>>(
      enc_w1, enc_w2, enc_w3, ne_w1, ne_w2, ne_w3);
  prep_edge_enc_w_kernel<<<(128 * 96 + 255) / 256, 256, 0, stream>>>(
      ee_w1, ee_w2, ee_w3, ee_w1p, ee_w2p, ee_w3p);
  prep_dec_w1_kernel<<<(96 * 128 + 255) / 256, 256, 0, stream>>>(dec_w1, dw1p);
  init_buffers_kernel<<<(N_NODES * HID + 255) / 256, 256, 0, stream>>>(
      outacc, mbuf, dbuf, N_NODES * HID, N_NODES);

  // ---- encoders ----
  node_encoder_kernel<<<(N_NODES + 31) / 32, 256, enc_smem, stream>>>(
      x, ne_w1, enc_b1, ne_w2, enc_b2, ne_w3, enc_b3, enc_lng, enc_lnb,
      y, ybf, N_NODES);
  edge_encoder_kernel<<<N_EDGES / 32, 256, enc_smem, stream>>>(
      eattr, ee_w1p, ee_b1, ee_w2p, ee_b2, ee_w3p, ee_b3, ee_lng, ee_lnb,
      eabf, N_EDGES);

  // ---- GATv2 layers ----
  for (int i = 0; i < NLAYERS; i++) {
    gemm128_kernel<<<N_NODES / 80, 256, 0, stream>>>(
        ybf, wlbf + (size_t)i * HH, gat_bl + i * HID, xl);
    gemm128_kernel<<<N_NODES / 80, 256, 0, stream>>>(
        ybf, wrbf + (size_t)i * HH, gat_br + i * HID, xr);
    edge_logits_kernel<<<N_EDGES / 64, 256, 0, stream>>>(
        eabf, webf + (size_t)i * HH, gat_att + i * HID, xl, xr, src, dst,
        sbuf, mbuf);
    edge_softmax_kernel<<<(N_EDGES + 255) / 256, 256, 0, stream>>>(
        sbuf, dst, mbuf, dbuf, zbuf, N_EDGES);
    edge_scatter_kernel<<<(N_EDGES * HID) / 256, 256, 0, stream>>>(
        zbuf, dbuf, xl, src, dst, outacc, N_EDGES * HID);
    node_update_kernel<<<(N_NODES + 255) / 256, 256, 0, stream>>>(
        y, ybf, outacc, gat_bias + i * HID, ln_g + i * HID, ln_b + i * HID,
        mbuf, dbuf, N_NODES);
  }

  // ---- decoder ----
  dec1_kernel<<<N_NODES / 80, 256, 0, stream>>>(ybf, dw1p, dec_b1, h1dec);
  dec23_kernel<<<(N_NODES + 255) / 256, 256, 0, stream>>>(
      h1dec, dec_w2, dec_b2, dec_w3, dec_b3, (float*)d_out, N_NODES);
}
